// HCOATT_TWC_15187004359016
// MI455X (gfx1250) — matrix operand order repaired and checked by reading, not compiled, not run on hardware
//
#include <hip/hip_runtime.h>
#include <hip/hip_bf16.h>
#include <math.h>

// ---------------- problem constants ----------------
#define BB 512
#define LL 20
#define NN_ 100
#define DD 64
#define EE 32
#define HH 3

typedef __attribute__((ext_vector_type(16))) _Float16 v16h;
typedef __attribute__((ext_vector_type(8)))  _Float16 v8h;
typedef __attribute__((ext_vector_type(8)))  float    v8f;
typedef __attribute__((ext_vector_type(4)))  unsigned int v4u;
typedef __attribute__((ext_vector_type(8)))  int      v8i;
typedef __attribute__((ext_vector_type(4)))  int      v4i;

// ---------------- TDM (Tensor Data Mover) helpers ----------------
#if defined(__has_builtin)
#if __has_builtin(__builtin_amdgcn_tensor_load_to_lds) && \
    __has_builtin(__builtin_amdgcn_s_wait_tensorcnt)
#define USE_TDM 1
#endif
#endif
#ifndef USE_TDM
#define USE_TDM 0
#endif

__device__ __forceinline__ unsigned int lds_off(const void* p) {
  // LDS aperture: addr[31:0] is the LDS byte offset (ISA 10.2 aperture table)
  return (unsigned int)(unsigned long long)p;
}

#if USE_TDM
// 2-D tile load: rows x width elements (elem size = 1<<dsz_code bytes), row
// stride `stride` elements, contiguous into LDS at lds_addr. D# per ISA ch.8.
// Toolchain uses the 6-arg builtin: (g0 v4u, g1 v8i, g2 v4i, g3 v4i, v8i, cpol).
__device__ __forceinline__ void tdm_load_2d(unsigned int lds_addr, const void* gptr,
                                            unsigned int dsz_code, unsigned int width,
                                            unsigned int rows, unsigned long long stride) {
  const unsigned long long ga = (unsigned long long)gptr;
  v4u g0;
  g0[0] = 1u;                                      // count = 1 (valid descriptor)
  g0[1] = lds_addr;                                // lds_addr      [63:32]
  g0[2] = (unsigned int)ga;                        // global_addr   [95:64]
  g0[3] = (unsigned int)(ga >> 32) | 0x80000000u;  // ga hi + type=2 [127:96]
  v8i g1;
  g1[0] = (int)(dsz_code << 16);                   // data_size [17:16]
  g1[1] = (int)(width << 16);                      // tensor_dim0 lo16 [63:48]
  g1[2] = (int)((width >> 16) | (rows << 16));     // dim0 hi | tensor_dim1 lo
  g1[3] = (int)((rows >> 16) | (width << 16));     // dim1 hi | tile_dim0
  g1[4] = (int)(rows & 0xFFFFu);                   // tile_dim1 (tile_dim2=0)
  g1[5] = (int)(unsigned int)stride;               // tensor_dim0_stride lo32
  g1[6] = (int)((stride >> 32) & 0xFFFFull);       // stride hi16 | dim1_stride lo16
  g1[7] = 0;
  const v4i z4 = {0, 0, 0, 0};
  const v8i z8 = {0, 0, 0, 0, 0, 0, 0, 0};
  __builtin_amdgcn_tensor_load_to_lds(g0, g1, z4, z4, z8, 0);
}
#endif

// ---------------- WMMA fragment loaders (CDNA5 wave32 layouts) ----------------
// A 16x32 f16 (MxK): per lane 16 contiguous k (k = hb*16 + s) -> two 128b loads.
__device__ __forceinline__ v16h load_a16(const _Float16* A, int lda, int row0,
                                         int kk, int nrows) {
  const int lane = threadIdx.x & 31;
  const int m = lane & 15, hb = lane >> 4;
  int r = row0 + m; if (r >= nrows) r = nrows - 1;
  const _Float16* p = A + (size_t)r * lda + kk + hb * 16;
  const v8h lo = *(const v8h*)(p);       // k = hb*16 .. hb*16+7
  const v8h hi = *(const v8h*)(p + 8);   // k = hb*16+8 .. hb*16+15
  v16h a;
#pragma unroll
  for (int i = 0; i < 8; ++i) { a[i] = lo[i]; a[8 + i] = hi[i]; }
  return a;
}

// B 32x16 f16 from N x K row-major ("NT"): 16 contiguous f16 per lane.
__device__ __forceinline__ v16h load_bT16(const _Float16* Bm, int ldb, int col0,
                                          int kk, int ncols) {
  const int lane = threadIdx.x & 31;
  const int n = lane & 15, hb = lane >> 4;
  int c = col0 + n; if (c >= ncols) c = ncols - 1;
  const _Float16* p = Bm + (size_t)c * ldb + kk + hb * 16; // lanes0-15: k0-15, 16-31: k16-31
  const v8h lo = *(const v8h*)(p);
  const v8h hi = *(const v8h*)(p + 8);
  v16h b;
#pragma unroll
  for (int i = 0; i < 8; ++i) { b[i] = lo[i]; b[8 + i] = hi[i]; }
  return b;
}

// "NT" B from f32 N x K row-major (LDS-staged weights), convert on load.
__device__ __forceinline__ v16h load_bT32(const float* W, int ldw, int col0, int kk) {
  const int lane = threadIdx.x & 31;
  const int n = lane & 15, hb = lane >> 4;
  const float* p = W + (size_t)(col0 + n) * ldw + kk + hb * 16;
  v16h b;
#pragma unroll
  for (int i = 0; i < 16; ++i) b[i] = (_Float16)p[i];
  return b;
}

// "NN" B from f32 K x N row-major (LDS-staged weights).
__device__ __forceinline__ v16h load_bN32(const float* W, int ldw, int col0, int kk) {
  const int lane = threadIdx.x & 31;
  const int n = lane & 15, hb = lane >> 4;
  const float* p = W + (size_t)(kk + hb * 16) * ldw + col0 + n;
  v16h b;
#pragma unroll
  for (int v = 0; v < 8; ++v) {
    b[2 * v]     = (_Float16)p[(2 * v) * ldw];
    b[2 * v + 1] = (_Float16)p[(2 * v + 1) * ldw];
  }
  return b;
}

// ---------------- kernels ----------------

// feature_embedding gather: (B,4,E)
__global__ void k_feat_embed(const int* uids, const int* iids, const int* tt, const int* tm,
                             const float* utab, const float* itab, const float* ttab,
                             const float* mtab, float* fe) {
  const int b = blockIdx.x, t = threadIdx.x;   // 128 threads
  const int row = t >> 5, e = t & 31;
  int id; const float* tab;
  if (row == 0)      { id = uids[b]; tab = utab; }
  else if (row == 1) { id = iids[b]; tab = itab; }
  else if (row == 2) { id = tt[b];   tab = ttab; }
  else               { id = tm[b];   tab = mtab; }
  fe[(size_t)b * 128 + row * 32 + e] = tab[(size_t)id * 32 + e];
}

// review gate: s = sum_n table[idx]; out = sigmoid(s@g1^T+b1)*tanh(s@g2^T+b2) -> f16
__global__ void k_review_gate(const int* rev_u, const int* rev_i,
                              const float* tab_u, const float* tab_i,
                              const float* g1w, const float* g1b,
                              const float* g2w, const float* g2b,
                              _Float16* out_u, _Float16* out_i) {
  __shared__ float s[DD];
  const int bl = blockIdx.x;         // 0..B*L-1
  const int which = blockIdx.y;      // 0 user, 1 item
  const int d = threadIdx.x;         // 64 threads
  const int* rv = (which ? rev_i : rev_u) + (size_t)bl * NN_;
  const float* tab = which ? tab_i : tab_u;
  float acc = 0.f;
  for (int n = 0; n < NN_; ++n) acc += tab[(size_t)rv[n] * DD + d];
  s[d] = acc;
  __syncthreads();
  float a1 = g1b[d], a2 = g2b[d];
  for (int k = 0; k < DD; ++k) { const float sv = s[k]; a1 += sv * g1w[d * DD + k]; a2 += sv * g2w[d * DD + k]; }
  const float val = (1.f / (1.f + expf(-a1))) * tanhf(a2);
  (which ? out_i : out_u)[(size_t)bl * DD + d] = (_Float16)val;
}

// C(Mx64,f16) = A(Mx64,f16) @ op(W 64x64 f32) + bias; transB=1 -> x@W^T, 0 -> x@W
// Block = 4 waves (wave = col tile), 4 row tiles per block; W staged in LDS via TDM.
#define RTPB 4
__global__ void k_gemm64(const _Float16* __restrict__ A, const float* __restrict__ W,
                         const float* __restrict__ bias, _Float16* __restrict__ C,
                         int transB, int Mrows) {
  __shared__ __align__(16) float Wsh[DD * DD];   // 16 KB
  const int wave = threadIdx.x >> 5;             // 0..3 -> column tile
  const int lane = threadIdx.x & 31;
#if USE_TDM
  if (wave == 0) {
    tdm_load_2d(lds_off(Wsh), W, /*4B*/2, DD, DD, DD);
    __builtin_amdgcn_s_wait_tensorcnt(0);
  }
#else
  for (int i = threadIdx.x; i < DD * DD; i += 128) Wsh[i] = W[i];
#endif
  __syncthreads();
  const int col = wave * 16 + (lane & 15);
  const int hb = lane >> 4;
  const float bv = bias ? bias[col] : 0.f;
  for (int rt = 0; rt < RTPB; ++rt) {
    const int row0 = (blockIdx.x * RTPB + rt) * 16;
    v8f acc = {};
#pragma unroll
    for (int kk = 0; kk < DD; kk += 32) {
      const v16h a = load_a16(A, DD, row0, kk, Mrows);
      const v16h b = transB ? load_bT32(Wsh, DD, wave * 16, kk)
                            : load_bN32(Wsh, DD, wave * 16, kk);
      acc = __builtin_amdgcn_wmma_f32_16x16x32_f16(false, a, false, b,
                                                   (short)0, acc, false, false);
    }
#pragma unroll
    for (int r = 0; r < 8; ++r)
      C[(size_t)(row0 + hb * 8 + r) * DD + col] = (_Float16)(acc[r] + bv);
  }
}

// batched S[b] = U[b](Rx64) @ V[b](Rx64)^T -> f32 (B,R,R).
// One block per batch; U[b],V[b] staged in LDS via TDM; 4 waves stride the tiles.
__global__ void k_score_gemm(const _Float16* __restrict__ U, const _Float16* __restrict__ V,
                             float* __restrict__ S, int R, int RT) {
  __shared__ __align__(16) _Float16 Ush[NN_ * DD];  // 12.8 KB
  __shared__ __align__(16) _Float16 Vsh[NN_ * DD];  // 12.8 KB
  const int b = blockIdx.x;
  const int wave = threadIdx.x >> 5, lane = threadIdx.x & 31;
  const _Float16* Ub = U + (size_t)b * R * DD;
  const _Float16* Vb = V + (size_t)b * R * DD;
#if USE_TDM
  if (wave == 0) {
    tdm_load_2d(lds_off(Ush), Ub, /*2B*/1, DD, (unsigned)R, DD);
    tdm_load_2d(lds_off(Vsh), Vb, /*2B*/1, DD, (unsigned)R, DD);
    __builtin_amdgcn_s_wait_tensorcnt(0);
  }
#else
  for (int i = threadIdx.x; i < R * DD; i += 128) { Ush[i] = Ub[i]; Vsh[i] = Vb[i]; }
#endif
  __syncthreads();
  float* Sb = S + (size_t)b * R * R;
  const int T = RT * RT;
  for (int t = wave; t < T; t += 4) {
    const int tu = t / RT, tv = t % RT;
    v8f acc = {};
#pragma unroll
    for (int kk = 0; kk < DD; kk += 32) {
      const v16h a  = load_a16(Ush, DD, tu * 16, kk, R);
      const v16h bf = load_bT16(Vsh, DD, tv * 16, kk, R);
      acc = __builtin_amdgcn_wmma_f32_16x16x32_f16(false, a, false, bf,
                                                   (short)0, acc, false, false);
    }
    const int colc = tv * 16 + (lane & 15);
    const int hb = lane >> 4;
    if (colc < R) {
#pragma unroll
      for (int r = 0; r < 8; ++r) {
        const int row = tu * 16 + hb * 8 + r;
        if (row < R) Sb[(size_t)row * R + colc] = acc[r];
      }
    }
  }
}

// row/col pooling of S: mode 0 = max, 1 = mean
__global__ void k_pool(const float* S, float* us, float* is_, int R, int mode) {
  const int b = blockIdx.x, t = threadIdx.x; // 128 threads
  if (t < R) {
    const float* Sb = S + (size_t)b * R * R;
    float mr = mode ? 0.f : -INFINITY, mc = mode ? 0.f : -INFINITY;
    for (int j = 0; j < R; ++j) {
      const float rv = Sb[(size_t)t * R + j];
      const float cv = Sb[(size_t)j * R + t];
      if (mode) { mr += rv; mc += cv; }
      else      { mr = fmaxf(mr, rv); mc = fmaxf(mc, cv); }
    }
    if (mode) { mr /= (float)R; mc /= (float)R; }
    us[b * R + t] = mr;
    is_[b * R + t] = mc;
  }
}

// hard gumbel forward == one_hot(argmax(score+g)); first-max tie-break
__global__ void k_argmax_gumbel(const float* score, const float* g, int* sel, int R) {
  const int b = blockIdx.x, t = threadIdx.x; // 32 threads
  float v = (t < R) ? score[b * R + t] + g[b * R + t] : -INFINITY;
  int idx = t;
  for (int off = 16; off > 0; off >>= 1) {
    const float ov = __shfl_down(v, off, 32);
    const int   oi = __shfl_down(idx, off, 32);
    if (ov > v || (ov == v && oi < idx)) { v = ov; idx = oi; }
  }
  if (t == 0) sel[b] = idx;
}

// words = table[reviews[b, sel[b], n]] -> f16 (B,N,64)
__global__ void k_gather_words(const int* reviews, const int* sel, const float* tab,
                               _Float16* out) {
  const int n = blockIdx.x, b = blockIdx.y, d = threadIdx.x; // 64 threads
  const int l = sel[b];
  const int w = reviews[((size_t)b * LL + l) * NN_ + n];
  out[((size_t)b * NN_ + n) * DD + d] = (_Float16)tab[(size_t)w * DD + d];
}

__global__ void k_softmax(const float* score, float* prob, int R) {
  __shared__ float red[128];
  const int b = blockIdx.x, t = threadIdx.x; // 128 threads
  const float v = (t < R) ? score[b * R + t] : -INFINITY;
  red[t] = v;
  __syncthreads();
  for (int off = 64; off > 0; off >>= 1) { if (t < off) red[t] = fmaxf(red[t], red[t + off]); __syncthreads(); }
  const float mx = red[0];
  __syncthreads();
  const float e = (t < R) ? expf(v - mx) : 0.f;
  red[t] = e;
  __syncthreads();
  for (int off = 64; off > 0; off >>= 1) { if (t < off) red[t] += red[t + off]; __syncthreads(); }
  if (t < R) prob[b * R + t] = e / red[0];
}

// fea[b,d] = sum_n words[b,n,d]*prob[b,n]  (reference uses u_words for BOTH sides)
__global__ void k_wsum(const _Float16* words, const float* pu, const float* pi,
                       float* feaU, float* feaI) {
  const int b = blockIdx.x, which = blockIdx.y, d = threadIdx.x; // 64 threads
  const float* p = (which ? pi : pu) + b * NN_;
  const _Float16* wb = words + (size_t)b * NN_ * DD + d;
  float acc = 0.f;
  for (int n = 0; n < NN_; ++n) acc += (float)wb[(size_t)n * DD] * p[n];
  (which ? feaI : feaU)[b * DD + d] = acc;
}

// ure/ire = relu(fea @ Wfc^T + b): (B,64)->(B,32), packed as rev_emb (B,2,32)
__global__ void k_small_fc(const float* feaU, const float* feaI,
                           const float* wu, const float* bu,
                           const float* wi, const float* bi, float* rev_emb) {
  __shared__ float f[DD];
  const int b = blockIdx.x, which = blockIdx.y, e = threadIdx.x; // 32 threads
  const float* fea = (which ? feaI : feaU) + (size_t)b * DD;
  f[e] = fea[e]; f[e + 32] = fea[e + 32];
  __syncthreads();
  const float* W = which ? wi : wu;
  float a = (which ? bi : bu)[e];
  for (int k = 0; k < DD; ++k) a += f[k] * W[e * DD + k];
  rev_emb[(size_t)b * 64 + which * 32 + e] = fmaxf(a, 0.f);
}

// feature-level co-attention (2xE vs 4xE), avg pooling, softmax atts
__global__ void k_feature_coatt(const float* rev_emb, const float* fe,
                                const float* Mf, const float* Wuf, const float* buf_,
                                const float* Wif, const float* bif,
                                float* u_feas, int h, int writeOut,
                                float* outU, float* outI) {
  __shared__ float rev[2][EE], fea[4][EE], u3[2][EE], i3[4][EE], uM3[2][EE];
  __shared__ float Sm[2][4], att[6];
  const int b = blockIdx.x, e = threadIdx.x; // 32 threads
  rev[0][e] = rev_emb[(size_t)b * 64 + e];
  rev[1][e] = rev_emb[(size_t)b * 64 + 32 + e];
  for (int r = 0; r < 4; ++r) fea[r][e] = fe[(size_t)b * 128 + r * 32 + e];
  __syncthreads();
  for (int r = 0; r < 2; ++r) {
    float a = buf_[e];
    for (int k = 0; k < EE; ++k) a += rev[r][k] * Wuf[e * EE + k];
    u3[r][e] = a;
  }
  for (int r = 0; r < 4; ++r) {
    float a = bif[e];
    for (int k = 0; k < EE; ++k) a += fea[r][k] * Wif[e * EE + k];
    i3[r][e] = a;
  }
  __syncthreads();
  for (int r = 0; r < 2; ++r) {
    float a = 0.f;
    for (int k = 0; k < EE; ++k) a += u3[r][k] * Mf[k * EE + e];
    uM3[r][e] = a;
  }
  __syncthreads();
  if (e < 8) {
    const int r = e >> 2, c = e & 3;
    float a = 0.f;
    for (int k = 0; k < EE; ++k) a += uM3[r][k] * i3[c][k];
    Sm[r][c] = a;
  }
  __syncthreads();
  if (e == 0) {
    float us0 = 0.f, us1 = 0.f, isc[4];
    for (int c = 0; c < 4; ++c) {
      us0 += Sm[0][c]; us1 += Sm[1][c];
      isc[c] = 0.5f * (Sm[0][c] + Sm[1][c]);
    }
    us0 *= 0.25f; us1 *= 0.25f;
    const float m = fmaxf(us0, us1);
    const float e0 = expf(us0 - m), e1 = expf(us1 - m);
    att[0] = e0 / (e0 + e1); att[1] = e1 / (e0 + e1);
    float mi = isc[0];
    for (int c = 1; c < 4; ++c) mi = fmaxf(mi, isc[c]);
    float se = 0.f, ex[4];
    for (int c = 0; c < 4; ++c) { ex[c] = expf(isc[c] - mi); se += ex[c]; }
    for (int c = 0; c < 4; ++c) att[2 + c] = ex[c] / se;
  }
  __syncthreads();
  u_feas[(size_t)b * 96 + h * 32 + e] = rev[0][e] * att[0];
  if (writeOut) { // last head defines the surviving feature rows
    outU[(size_t)b * 96 + e]      = fea[0][e] * att[2]; // user_id_e
    outU[(size_t)b * 96 + 32 + e] = fea[2][e] * att[4]; // type_e
    outI[(size_t)b * 96 + e]      = fea[1][e] * att[3]; // item_id_e
    outI[(size_t)b * 96 + 32 + e] = fea[3][e] * att[5]; // month_e
  }
}

// u_out/i_out = concat(u_feas) @ fc^T + b -> row 2 of each output stack
__global__ void k_final(const float* u_feas, const float* uw, const float* ub,
                        const float* iw, const float* ib, float* outU, float* outI) {
  __shared__ float f[96];
  const int b = blockIdx.x, e = threadIdx.x; // 32 threads
  f[e] = u_feas[(size_t)b * 96 + e];
  f[e + 32] = u_feas[(size_t)b * 96 + 32 + e];
  f[e + 64] = u_feas[(size_t)b * 96 + 64 + e];
  __syncthreads();
  float au = ub[e], ai = ib[e];
  for (int k = 0; k < 96; ++k) { au += f[k] * uw[e * 96 + k]; ai += f[k] * iw[e * 96 + k]; }
  outU[(size_t)b * 96 + 64 + e] = au;
  outI[(size_t)b * 96 + 64 + e] = ai;
}

// ---------------- launch ----------------
extern "C" void kernel_launch(void* const* d_in, const int* in_sizes, int n_in,
                              void* d_out, int out_size, void* d_ws, size_t ws_size,
                              hipStream_t stream) {
  (void)in_sizes; (void)n_in; (void)out_size; (void)ws_size;
  const int*   user_reviews   = (const int*)d_in[0];
  const int*   item_reviews   = (const int*)d_in[1];
  const int*   uids           = (const int*)d_in[2];
  const int*   iids           = (const int*)d_in[3];
  const int*   travel_type    = (const int*)d_in[4];
  const int*   travel_month   = (const int*)d_in[5];
  const float* gnoise_u       = (const float*)d_in[6];
  const float* gnoise_i       = (const float*)d_in[7];
  const float* user_word_embs = (const float*)d_in[8];
  const float* item_word_embs = (const float*)d_in[9];
  const float* user_id_table  = (const float*)d_in[10];
  const float* item_id_table  = (const float*)d_in[11];
  const float* type_table     = (const float*)d_in[12];
  const float* month_table    = (const float*)d_in[13];
  const float* fc_g1_w = (const float*)d_in[14];
  const float* fc_g1_b = (const float*)d_in[15];
  const float* fc_g2_w = (const float*)d_in[16];
  const float* fc_g2_b = (const float*)d_in[17];
  const float* M_r  = (const float*)d_in[18];
  const float* Wu_r = (const float*)d_in[19];
  const float* bu_r = (const float*)d_in[20];
  const float* Wi_r = (const float*)d_in[21];
  const float* bi_r = (const float*)d_in[22];
  const float* M_w  = (const float*)d_in[23];
  const float* Wu_w = (const float*)d_in[24];
  const float* bu_w = (const float*)d_in[25];
  const float* Wi_w = (const float*)d_in[26];
  const float* bi_w = (const float*)d_in[27];
  const float* M_f  = (const float*)d_in[28];
  const float* Wu_f = (const float*)d_in[29];
  const float* bu_f = (const float*)d_in[30];
  const float* Wi_f = (const float*)d_in[31];
  const float* bi_f = (const float*)d_in[32];
  const float* ufc_w = (const float*)d_in[33];
  const float* ufc_b = (const float*)d_in[34];
  const float* ifc_w = (const float*)d_in[35];
  const float* ifc_b = (const float*)d_in[36];
  const float* u_fc_w = (const float*)d_in[37];
  const float* u_fc_b = (const float*)d_in[38];
  const float* i_fc_w = (const float*)d_in[39];
  const float* i_fc_b = (const float*)d_in[40];

  float* outU = (float*)d_out;
  float* outI = outU + (size_t)BB * 96;

  // workspace carve-up (~62 MB total, reused across heads)
  char* p = (char*)d_ws;
  auto alloc = [&](size_t bytes) -> char* {
    char* r = p; p += (bytes + 255) & ~(size_t)255; return r;
  };
  float*    feat_emb = (float*)   alloc((size_t)BB * 4 * EE * 4);
  _Float16* urg      = (_Float16*)alloc((size_t)BB * LL * DD * 2);
  _Float16* irg      = (_Float16*)alloc((size_t)BB * LL * DD * 2);
  _Float16* uWr      = (_Float16*)alloc((size_t)BB * LL * DD * 2);
  _Float16* iWr      = (_Float16*)alloc((size_t)BB * LL * DD * 2);
  _Float16* uMr      = (_Float16*)alloc((size_t)BB * LL * DD * 2);
  float*    S_r      = (float*)   alloc((size_t)BB * LL * LL * 4);
  int*      sel_u    = (int*)     alloc((size_t)BB * 4);
  int*      sel_i    = (int*)     alloc((size_t)BB * 4);
  _Float16* words_u  = (_Float16*)alloc((size_t)BB * NN_ * DD * 2);
  _Float16* words_i  = (_Float16*)alloc((size_t)BB * NN_ * DD * 2);
  _Float16* uWw      = (_Float16*)alloc((size_t)BB * NN_ * DD * 2);
  _Float16* iWw      = (_Float16*)alloc((size_t)BB * NN_ * DD * 2);
  _Float16* uMw      = (_Float16*)alloc((size_t)BB * NN_ * DD * 2);
  float*    S_w      = (float*)   alloc((size_t)BB * NN_ * NN_ * 4);
  float*    score_u  = (float*)   alloc((size_t)BB * NN_ * 4);
  float*    score_i  = (float*)   alloc((size_t)BB * NN_ * 4);
  float*    prob_u   = (float*)   alloc((size_t)BB * NN_ * 4);
  float*    prob_i   = (float*)   alloc((size_t)BB * NN_ * 4);
  float*    u_w_fea  = (float*)   alloc((size_t)BB * DD * 4);
  float*    i_w_fea  = (float*)   alloc((size_t)BB * DD * 4);
  float*    rev_emb  = (float*)   alloc((size_t)BB * 2 * EE * 4);
  float*    u_feas   = (float*)   alloc((size_t)BB * 96 * 4);

  // stage 0: embeddings + gated review features
  k_feat_embed<<<BB, 128, 0, stream>>>(uids, iids, travel_type, travel_month,
                                       user_id_table, item_id_table, type_table,
                                       month_table, feat_emb);
  k_review_gate<<<dim3(BB * LL, 2), 64, 0, stream>>>(
      user_reviews, item_reviews, user_word_embs, item_word_embs,
      fc_g1_w, fc_g1_b, fc_g2_w, fc_g2_b, urg, irg);

  const int MR = BB * LL;   // 10240 rows -> 160 blocks of 4 row tiles
  const int MW = BB * NN_;  // 51200 rows -> 800 blocks of 4 row tiles

  for (int h = 0; h < HH; ++h) {
    // ---- review-level co-attention (max pool + gumbel argmax) ----
    k_gemm64<<<MR / (16 * RTPB), 128, 0, stream>>>(urg, Wu_r + h * DD * DD, bu_r + h * DD, uWr, 1, MR);
    k_gemm64<<<MR / (16 * RTPB), 128, 0, stream>>>(irg, Wi_r + h * DD * DD, bi_r + h * DD, iWr, 1, MR);
    k_gemm64<<<MR / (16 * RTPB), 128, 0, stream>>>(uWr, M_r + h * DD * DD, nullptr, uMr, 0, MR);
    k_score_gemm<<<BB, 128, 0, stream>>>(uMr, iWr, S_r, LL, 2);
    k_pool<<<BB, 128, 0, stream>>>(S_r, score_u, score_i, LL, 0);
    k_argmax_gumbel<<<BB, 32, 0, stream>>>(score_u, gnoise_u + (size_t)h * BB * LL, sel_u, LL);
    k_argmax_gumbel<<<BB, 32, 0, stream>>>(score_i, gnoise_i + (size_t)h * BB * LL, sel_i, LL);

    // ---- word gather + word-level co-attention (mean pool + softmax) ----
    k_gather_words<<<dim3(NN_, BB), 64, 0, stream>>>(user_reviews, sel_u, user_word_embs, words_u);
    k_gather_words<<<dim3(NN_, BB), 64, 0, stream>>>(item_reviews, sel_i, item_word_embs, words_i);
    k_gemm64<<<MW / (16 * RTPB), 128, 0, stream>>>(words_u, Wu_w + h * DD * DD, bu_w + h * DD, uWw, 1, MW);
    k_gemm64<<<MW / (16 * RTPB), 128, 0, stream>>>(words_i, Wi_w + h * DD * DD, bi_w + h * DD, iWw, 1, MW);
    k_gemm64<<<MW / (16 * RTPB), 128, 0, stream>>>(uWw, M_w + h * DD * DD, nullptr, uMw, 0, MW);
    k_score_gemm<<<BB, 128, 0, stream>>>(uMw, iWw, S_w, NN_, 7);
    k_pool<<<BB, 128, 0, stream>>>(S_w, score_u, score_i, NN_, 1);
    k_softmax<<<BB, 128, 0, stream>>>(score_u, prob_u, NN_);
    k_softmax<<<BB, 128, 0, stream>>>(score_i, prob_i, NN_);
    k_wsum<<<dim3(BB, 2), 64, 0, stream>>>(words_u, prob_u, prob_i, u_w_fea, i_w_fea);
    k_small_fc<<<dim3(BB, 2), 32, 0, stream>>>(u_w_fea, i_w_fea, ufc_w, ufc_b, ifc_w, ifc_b, rev_emb);

    // ---- feature-level co-attention ----
    k_feature_coatt<<<BB, 32, 0, stream>>>(
        rev_emb, feat_emb, M_f + h * EE * EE, Wu_f + h * EE * EE, bu_f + h * EE,
        Wi_f + h * EE * EE, bi_f + h * EE, u_feas, h, (h == HH - 1) ? 1 : 0, outU, outI);
  }

  k_final<<<BB, 32, 0, stream>>>(u_feas, u_fc_w, u_fc_b, i_fc_w, i_fc_b, outU, outI);
}
